// MemorySelfAttention_65171833750055
// MI455X (gfx1250) — compile-verified, matrix operand order checked by reading, not run on hardware
//
#include <hip/hip_runtime.h>

typedef __bf16 bf16;
typedef bf16  v16bf __attribute__((ext_vector_type(16)));
typedef float v8f   __attribute__((ext_vector_type(8)));

#define HD 128

__device__ inline bf16 f2bf(float f) {
  union { float f; unsigned u; } a; a.f = f;
  unsigned r = a.u + 0x7FFFu + ((a.u >> 16) & 1u);   // RNE
  unsigned short h = (unsigned short)(r >> 16);
  bf16 b; __builtin_memcpy(&b, &h, 2);
  return b;
}
__device__ inline float bf2f(bf16 b) {
  unsigned short u; __builtin_memcpy(&u, &b, 2);
  unsigned v = ((unsigned)u) << 16;
  float f; __builtin_memcpy(&f, &v, 4);
  return f;
}

__device__ inline v8f wmma_bf16(v16bf a, v16bf b, v8f c) {
  return __builtin_amdgcn_wmma_f32_16x16x32_bf16(false, a, false, b, (short)0, c, false, false);
}

// Load A/B operand: two contiguous 8-element (16B) chunks at p and p+16.
__device__ inline v16bf load_ab(const bf16* p) {
  v16bf r;
#pragma unroll
  for (int i = 0; i < 8; i++) r[i] = p[i];
#pragma unroll
  for (int i = 0; i < 8; i++) r[8 + i] = p[16 + i];
  return r;
}

// ---------------- elementwise kernels ----------------

__global__ void k_cvt(const float* __restrict__ s, bf16* __restrict__ d, long n) {
  long i = (long)blockIdx.x * blockDim.x + threadIdx.x;
  if (i < n) d[i] = f2bf(s[i]);
}

// seq = concat(stm[B,256,2048], x[B,1024,2048]) -> bf16 [B,1280,2048]
__global__ void k_pack_seq(const float* __restrict__ stm, const float* __restrict__ x,
                           bf16* __restrict__ seq, long n) {
  long i = (long)blockIdx.x * blockDim.x + threadIdx.x;
  if (i >= n) return;
  int c = (int)(i & 2047);
  long ri = i >> 11;
  int row = (int)(ri % 1280);
  int b = (int)(ri / 1280);
  float v = (row < 256) ? stm[((long)b * 256 + row) * 2048 + c]
                        : x[((long)b * 1024 + (row - 256)) * 2048 + c];
  seq[i] = f2bf(v);
}

// long_k/long_v [B,512,8,128] f32 -> Kf/Vf [B,8,1792,128] bf16 rows 0..511
__global__ void k_ldkv(const float* __restrict__ lk, const float* __restrict__ lv,
                       bf16* __restrict__ Kf, bf16* __restrict__ Vf, long n) {
  long i = (long)blockIdx.x * blockDim.x + threadIdx.x;
  if (i >= n) return;
  int d = (int)(i & 127);
  long r1 = i >> 7;
  int h = (int)(r1 & 7);
  long r2 = r1 >> 3;
  int row = (int)(r2 % 512);
  int b = (int)(r2 / 512);
  long dst = (((long)(b * 8 + h)) * 1792 + row) * HD + d;
  Kf[dst] = f2bf(lk[i]);
  Vf[dst] = f2bf(lv[i]);
}

// In-place interleaved RoPE on bf16 planes
__global__ void k_rope(bf16* __restrict__ X, long planeStride, int rowStart,
                       int rows, int posBase, long n) {
  long i = (long)blockIdx.x * blockDim.x + threadIdx.x;
  if (i >= n) return;
  int j = (int)(i & 63);                 // pair index 0..63
  long r1 = i >> 6;
  int r = (int)(r1 % rows);
  long pl = r1 / rows;
  bf16* p = X + pl * planeStride + (long)(rowStart + r) * HD + 2 * j;
  float x1 = bf2f(p[0]), x2 = bf2f(p[1]);
  float invf = __powf(10000.0f, -(float)j / 64.0f);
  float ang = (float)(posBase + r) * invf;
  float c = __cosf(ang), s = __sinf(ang);
  p[0] = f2bf(x1 * c - x2 * s);
  p[1] = f2bf(x1 * s + x2 * c);
}

// ---------------- WMMA GEMM: C[b,M,N] = A[b,M,K] * Bw[N,K]^T ----------------
// Block 256 = 8 waves, tile 32(M) x 256(N); wave tile 16x64; K steps of 32.
template <int F32OUT, int HEADSPLIT>
__global__ __launch_bounds__(256) void k_gemm(const bf16* __restrict__ A,
                                              const bf16* __restrict__ Bw,
                                              void* __restrict__ Cout,
                                              int M, int N, int Kd,
                                              long sAb, long sCb,
                                              int rowsTot, int rowOff) {
  const int lane = threadIdx.x & 31;
  const int wave = threadIdx.x >> 5;
  const int bz = blockIdx.z;
  const int mbase = blockIdx.y * 32 + (wave >> 2) * 16;
  const int nbase = blockIdx.x * 256 + (wave & 3) * 64;
  const int rrow = lane & 15;
  const int koff = (lane < 16) ? 0 : 8;
  const bf16* Ap = A + bz * sAb + (long)(mbase + rrow) * Kd;

  v8f acc[4];
#pragma unroll
  for (int j = 0; j < 4; j++)
#pragma unroll
    for (int e = 0; e < 8; e++) acc[j][e] = 0.0f;

  for (int k = 0; k < Kd; k += 32) {
    // warm the next K-chunk lines (lowers to global_prefetch_b8)
    if (k + 32 < Kd) {
      __builtin_prefetch(Ap + k + 32, 0, 0);
      __builtin_prefetch(Bw + (long)(nbase + rrow) * Kd + k + 32, 0, 0);
    }
    v16bf a = load_ab(Ap + k + koff);
    v16bf bb[4];
#pragma unroll
    for (int j = 0; j < 4; j++)
      bb[j] = load_ab(Bw + (long)(nbase + j * 16 + rrow) * Kd + k + koff);
#pragma unroll
    for (int j = 0; j < 4; j++) acc[j] = wmma_bf16(a, bb[j], acc[j]);
  }

#pragma unroll
  for (int j = 0; j < 4; j++) {
#pragma unroll
    for (int v = 0; v < 8; v++) {
      int row = mbase + v + 8 * (lane >> 4);
      int n = nbase + j * 16 + rrow;
      long off;
      if (HEADSPLIT)
        off = bz * sCb + (long)(n >> 7) * rowsTot * HD + (long)(rowOff + row) * HD + (n & 127);
      else
        off = bz * sCb + (long)row * N + n;
      if (F32OUT) ((float*)Cout)[off] = acc[j][v];
      else        ((bf16*)Cout)[off] = f2bf(acc[j][v]);
    }
  }
}

// ---------------- Flash attention (GQA, causal offset 768) ----------------
// Q [B,16,1024,128] bf16 (post-RoPE), K/V [B,8,1792,128] bf16.
// O [B,1024,2048] bf16.  grid: x = B*16, y = 16 (64-query tiles). block = 128.
// Register-staged pipeline: global loads for tile t+1 are issued before the
// WMMA compute of tile t, hiding HBM/L2 latency behind matrix math.
__global__ __launch_bounds__(128) void k_flash(const bf16* __restrict__ Q,
                                               const bf16* __restrict__ K,
                                               const bf16* __restrict__ V,
                                               bf16* __restrict__ O) {
  __shared__ bf16 sK[32 * 136];       // [key][dim], padded (16B-aligned rows)
  __shared__ bf16 sV[HD * 40];        // [dim][key], padded
  __shared__ bf16 sP[4 * 16 * 40];    // per-wave P bounce [row][key]

  const int lane = threadIdx.x & 31;
  const int wave = threadIdx.x >> 5;
  const int bh = blockIdx.x;
  const int b = bh >> 4, h = bh & 15;
  const int kvh = h >> 1;
  const int qb = blockIdx.y * 64;
  const int qr = qb + wave * 16;

  const bf16* Qp = Q + (((long)b * 16 + h) * 1024 + qr) * HD;
  const bf16* Kp = K + (((long)b * 8 + kvh) * 1792) * HD;
  const bf16* Vp = V + (((long)b * 8 + kvh) * 1792) * HD;

  const int rrow = lane & 15;
  const int koff = (lane < 16) ? 0 : 8;
  const int mrow0 = 8 * (lane >> 4);
  const float scale = 0.08838834764831845f;  // 1/sqrt(128)

  // Preload Q rows as A operands (4 dim-chunks of 32)
  v16bf qa[4];
#pragma unroll
  for (int c = 0; c < 4; c++) qa[c] = load_ab(Qp + rrow * HD + c * 32 + koff);

  v8f acco[8];
#pragma unroll
  for (int t = 0; t < 8; t++)
#pragma unroll
    for (int e = 0; e < 8; e++) acco[t][e] = 0.0f;
  float mrun[8], lrun[8];
#pragma unroll
  for (int v = 0; v < 8; v++) { mrun[v] = -1e30f; lrun[v] = 0.0f; }

  const int nt = (qb + 832) / 32;     // key tiles for this block
  bf16* pw = &sP[wave * 16 * 40];

  // Per-thread staging slots: 4x 16B chunks each for K and V.
  union U { uint4 u; bf16 h[8]; };
  U kreg[4], vreg[4];
#pragma unroll
  for (int j = 0; j < 4; j++) {
    int i = threadIdx.x + j * 128;
    int kk = i >> 4, d8 = (i & 15) * 8;
    kreg[j].u = *(const uint4*)(Kp + (long)kk * HD + d8);
    vreg[j].u = *(const uint4*)(Vp + (long)kk * HD + d8);
  }

  for (int t = 0; t < nt; t++) {
    if (t) __syncthreads();           // all waves done reading previous tile
    // Commit staged registers to LDS (K row-major, V transposed dim-major)
#pragma unroll
    for (int j = 0; j < 4; j++) {
      int i = threadIdx.x + j * 128;
      int kk = i >> 4, d8 = (i & 15) * 8;
      *(uint4*)&sK[kk * 136 + d8] = kreg[j].u;
#pragma unroll
      for (int e = 0; e < 8; e++) sV[(d8 + e) * 40 + kk] = vreg[j].h[e];
    }
    __syncthreads();

    // Issue global loads for tile t+1 (in flight during compute below)
    if (t + 1 < nt) {
      long kb0 = (long)(t + 1) * 32;
#pragma unroll
      for (int j = 0; j < 4; j++) {
        int i = threadIdx.x + j * 128;
        int kk = i >> 4, d8 = (i & 15) * 8;
        kreg[j].u = *(const uint4*)(Kp + (kb0 + kk) * HD + d8);
        vreg[j].u = *(const uint4*)(Vp + (kb0 + kk) * HD + d8);
      }
    }

    const int b0 = t * 32;

    // S = Q * K^T : preload all B operands, then WMMA chain
    v16bf kb[8];
#pragma unroll
    for (int n16 = 0; n16 < 2; n16++)
#pragma unroll
      for (int c = 0; c < 4; c++)
        kb[n16 * 4 + c] = load_ab(&sK[(n16 * 16 + rrow) * 136 + c * 32 + koff]);

    v8f s[2];
#pragma unroll
    for (int u = 0; u < 2; u++)
#pragma unroll
      for (int e = 0; e < 8; e++) s[u][e] = 0.0f;
#pragma unroll
    for (int n16 = 0; n16 < 2; n16++)
#pragma unroll
      for (int c = 0; c < 4; c++)
        s[n16] = wmma_bf16(qa[c], kb[n16 * 4 + c], s[n16]);

    // scale + causal mask (key <= row + 768), tile row-max
    float tmax[8];
#pragma unroll
    for (int v = 0; v < 8; v++) {
      int r = qr + mrow0 + v;
      float a0 = s[0][v] * scale;
      float a1 = s[1][v] * scale;
      if (b0 + rrow > r + 768) a0 = -1e30f;
      if (b0 + 16 + rrow > r + 768) a1 = -1e30f;
      s[0][v] = a0; s[1][v] = a1;
      tmax[v] = fmaxf(a0, a1);
    }
#pragma unroll
    for (int off = 1; off < 16; off <<= 1)
#pragma unroll
      for (int v = 0; v < 8; v++)
        tmax[v] = fmaxf(tmax[v], __shfl_xor(tmax[v], off, 32));

    // online softmax update; write P (bf16) to per-wave LDS bounce
    float psum[8];
#pragma unroll
    for (int v = 0; v < 8; v++) {
      float mnew = fmaxf(mrun[v], tmax[v]);
      float corr = __expf(mrun[v] - mnew);
      mrun[v] = mnew;
      lrun[v] *= corr;
#pragma unroll
      for (int u = 0; u < 8; u++) acco[u][v] = acco[u][v] * corr;
      float p0 = __expf(s[0][v] - mnew);
      float p1 = __expf(s[1][v] - mnew);
      psum[v] = p0 + p1;
      int m = mrow0 + v;
      pw[m * 40 + rrow] = f2bf(p0);
      pw[m * 40 + 16 + rrow] = f2bf(p1);
    }
#pragma unroll
    for (int off = 1; off < 16; off <<= 1)
#pragma unroll
      for (int v = 0; v < 8; v++) psum[v] += __shfl_xor(psum[v], off, 32);
#pragma unroll
    for (int v = 0; v < 8; v++) lrun[v] += psum[v];

    // Reload P in A-layout (wave-local LDS: DS ops are in-order per wave)
    v16bf pa = load_ab(&pw[rrow * 40 + koff]);

    // O += P * V : preload all B operands, then WMMA chain
    v16bf vb[8];
#pragma unroll
    for (int u = 0; u < 8; u++)
      vb[u] = load_ab(&sV[(u * 16 + rrow) * 40 + koff]);
#pragma unroll
    for (int u = 0; u < 8; u++)
      acco[u] = wmma_bf16(pa, vb[u], acco[u]);
  }

  // normalize + store O[b, r, h*128 + d] as bf16
#pragma unroll
  for (int v = 0; v < 8; v++) {
    float inv = 1.0f / lrun[v];
    int m = mrow0 + v;
    int r = qr + m;
    bf16* op = O + ((long)b * 1024 + r) * 2048 + h * HD;
#pragma unroll
    for (int u = 0; u < 8; u++) op[u * 16 + rrow] = f2bf(acco[u][v] * inv);
  }
}

// ---------------- host launch ----------------

static inline int gblocks(long n) { return (int)((n + 255) / 256); }

extern "C" void kernel_launch(void* const* d_in, const int* in_sizes, int n_in,
                              void* d_out, int out_size, void* d_ws, size_t ws_size,
                              hipStream_t stream) {
  const float* x   = (const float*)d_in[0];
  const float* stm = (const float*)d_in[1];
  const float* lk  = (const float*)d_in[2];
  const float* lv  = (const float*)d_in[3];
  const float* Wq  = (const float*)d_in[4];
  const float* Wk  = (const float*)d_in[5];
  const float* Wv  = (const float*)d_in[6];
  const float* Wo  = (const float*)d_in[7];
  float* out = (float*)d_out;

  char* ws = (char*)d_ws;
  size_t off = 0;
  bf16* seq = (bf16*)(ws + off); off += (size_t)4 * 1280 * 2048 * 2;
  bf16* wq  = (bf16*)(ws + off); off += (size_t)2048 * 2048 * 2;
  bf16* wk  = (bf16*)(ws + off); off += (size_t)1024 * 2048 * 2;
  bf16* wv  = (bf16*)(ws + off); off += (size_t)1024 * 2048 * 2;
  bf16* wo  = (bf16*)(ws + off); off += (size_t)2048 * 2048 * 2;
  bf16* Qh  = (bf16*)(ws + off); off += (size_t)4 * 16 * 1024 * 128 * 2;
  bf16* Kf  = (bf16*)(ws + off); off += (size_t)4 * 8 * 1792 * 128 * 2;
  bf16* Vf  = (bf16*)(ws + off); off += (size_t)4 * 8 * 1792 * 128 * 2;
  bf16* Ob  = (bf16*)(ws + off); off += (size_t)4 * 1024 * 2048 * 2;

  {
    long n = (long)4 * 1280 * 2048;
    k_pack_seq<<<gblocks(n), 256, 0, stream>>>(stm, x, seq, n);
  }
  k_cvt<<<gblocks((long)2048 * 2048), 256, 0, stream>>>(Wq, wq, (long)2048 * 2048);
  k_cvt<<<gblocks((long)1024 * 2048), 256, 0, stream>>>(Wk, wk, (long)1024 * 2048);
  k_cvt<<<gblocks((long)1024 * 2048), 256, 0, stream>>>(Wv, wv, (long)1024 * 2048);
  k_cvt<<<gblocks((long)2048 * 2048), 256, 0, stream>>>(Wo, wo, (long)2048 * 2048);
  {
    long n = (long)4 * 512 * 8 * 128;
    k_ldkv<<<gblocks(n), 256, 0, stream>>>(lk, lv, Kf, Vf, n);
  }
  // Q projection: last 1024 rows of seq -> [B,16,1024,128]
  k_gemm<0, 1><<<dim3(2048 / 256, 1024 / 32, 4), 256, 0, stream>>>(
      seq + (size_t)256 * 2048, wq, Qh, 1024, 2048, 2048,
      (long)1280 * 2048, (long)16 * 1024 * 128, 1024, 0);
  // K projection: all 1280 rows -> [B,8,1792,128] at row 512
  k_gemm<0, 1><<<dim3(1024 / 256, 1280 / 32, 4), 256, 0, stream>>>(
      seq, wk, Kf, 1280, 1024, 2048,
      (long)1280 * 2048, (long)8 * 1792 * 128, 1792, 512);
  // V projection
  k_gemm<0, 1><<<dim3(1024 / 256, 1280 / 32, 4), 256, 0, stream>>>(
      seq, wv, Vf, 1280, 1024, 2048,
      (long)1280 * 2048, (long)8 * 1792 * 128, 1792, 512);
  // RoPE: Q positions 256..1279, new K rows positions 0..1279
  {
    long n = (long)4 * 16 * 1024 * 64;
    k_rope<<<gblocks(n), 256, 0, stream>>>(Qh, (long)1024 * 128, 0, 1024, 256, n);
  }
  {
    long n = (long)4 * 8 * 1280 * 64;
    k_rope<<<gblocks(n), 256, 0, stream>>>(Kf, (long)1792 * 128, 512, 1280, 0, n);
  }
  // flash attention
  k_flash<<<dim3(4 * 16, 1024 / 64), 128, 0, stream>>>(Qh, Kf, Vf, Ob);
  // output projection -> f32 d_out [B,1024,2048]
  k_gemm<1, 0><<<dim3(2048 / 256, 1024 / 32, 4), 256, 0, stream>>>(
      Ob, wo, out, 1024, 2048, 2048,
      (long)1024 * 2048, (long)1024 * 2048, 0, 0);
  (void)in_sizes; (void)n_in; (void)out_size; (void)ws_size;
}
